// FourierKANLayer_62277025792698
// MI455X (gfx1250) — matrix operand order repaired and checked by reading, not compiled, not run on hardware
//
#include <hip/hip_runtime.h>

// ---------------------------------------------------------------------------
// FourierKAN as a WMMA GEMM on gfx1250 (MI455X).
//   y[b,f] = sum_{i,g} cos((g+1)*x[b,i]) * W0[f,i,g] + sin((g+1)*x[b,i]) * W1[f,i,g]
// == (4096 x 8192) * (8192 x 512) GEMM with K index kk = i*16 + trig*8 + g.
// Compute-bound (34.4 GFLOP, 32 MB footprint -> fits in 192MB L2), so we use
// v_wmma_f32_16x16x32_f16 with f32 accumulation.
// ---------------------------------------------------------------------------

typedef __attribute__((ext_vector_type(16))) _Float16 v16h;
typedef __attribute__((ext_vector_type(8)))  _Float16 v8h;
typedef __attribute__((ext_vector_type(8)))  float    v8f;
typedef __attribute__((ext_vector_type(4)))  float    v4f;

#define BATCH    4096
#define IN_DIM   512
#define FEATURES 512
#define GRID_N   8
// K = IN_DIM * GRID_N * 2 = 8192 ; processed in 256 steps of 32.

// ---------------------------------------------------------------------------
// Pass 1: prepack weights f32 -> f16 in a "fragment ready" K x F layout.
//
// WMMA B fragment (16-bit, 32x16 KxN): lane n (0..15) half h (0/1) is given the
// 16 f16 values B[K = h*16 + 0..15][n] contiguously (contiguous k order, same as A; pair-packed per VGPR).
// With kk = i*16 + trig*8 + g and K-step t covering i = 2t, 2t+1, the lane's
// values for (t, f, h) are:
//    j<8 : kernel[0][f][2t+h][j]   (cos weights, g = j+1)
//    j>=8: kernel[1][f][2t+h][j-8] (sin weights)
// Stored at wsB[ ((t*512 + f)*2 + h)*16 + j ].
// ---------------------------------------------------------------------------
__global__ __launch_bounds__(256) void fkan_prepack_w(
    const float* __restrict__ kern, _Float16* __restrict__ wsB) {
  // One thread per source chunk of 8 contiguous floats (fully coalesced reads).
  // s = trig*(512*512) + f*512 + (2t+h)   (chunk units of 8 floats)
  const int s    = blockIdx.x * 256 + threadIdx.x;      // 0 .. 524287
  const int trig = s >> 18;                             // 512*512 = 2^18
  const int f    = (s >> 9) & 511;
  const int ih   = s & 511;                             // = 2t + h
  const int t    = ih >> 1;
  const int h    = ih & 1;

  const float* src = kern + (size_t)trig * (512u * 512u * 8u)
                          + (size_t)f    * (512u * 8u)
                          + (size_t)ih   * 8u;
  v4f a = *(const v4f*)(src);
  v4f b = *(const v4f*)(src + 4);

  v8h o;
  o[0] = (_Float16)a[0]; o[1] = (_Float16)a[1];
  o[2] = (_Float16)a[2]; o[3] = (_Float16)a[3];
  o[4] = (_Float16)b[0]; o[5] = (_Float16)b[1];
  o[6] = (_Float16)b[2]; o[7] = (_Float16)b[3];

  const size_t e = (size_t)t * 16384u + (size_t)f * 32u + (size_t)h * 16u
                 + (size_t)trig * 8u;                   // f16 element index
  *(v8h*)(wsB + e) = o;                                  // 16B aligned store
}

// ---------------------------------------------------------------------------
// Pass 2: the WMMA GEMM. Block = 256 threads = 8 wave32s; block tile 128x128.
// Wave w owns rows [bx*128 + 16w, +16) and all 128 block columns via 8
// 16x16 f32 accumulators. A-fragments are synthesized per lane with one
// v_cos_f32 per element: sin(p) = cos(p - pi/2), in elements 8..15 of every lane.
// ---------------------------------------------------------------------------
__global__ __launch_bounds__(256) void fkan_wmma_gemm(
    const float* __restrict__ x, const _Float16* __restrict__ wsB,
    float* __restrict__ y) {
  const int wave = threadIdx.x >> 5;
  const int lane = threadIdx.x & 31;
  const int lh   = lane >> 4;   // lane-half: 0 => i = 2t / K 0..15, 1 => i = 2t+1 / K 16..31
  const int ln   = lane & 15;

  const int row = blockIdx.x * 128 + wave * 16 + ln;    // A-matrix row (M = ln)
  const int f0  = blockIdx.y * 128;                     // feature tile base

  const float2* xrow = (const float2*)(x + (size_t)row * IN_DIM);
  const float   bias = lh ? -1.57079632679489662f : 0.0f;  // sin = cos(p - pi/2)

  v8f acc[8] = {};  // 8 x (16x16 f32) accumulators = 16 rows x 128 cols

  for (int t = 0; t < 256; ++t) {
    // --- A fragment: lane-half lh needs i = 2t+lh, elements 0..7 cos (trig 0)
    // and 8..15 sin (trig 1), g = 1..8, matching the prepacked B order.
    const float2 x01 = xrow[t];
    const float xi = lh ? x01.y : x01.x;
    v16h a;
#pragma unroll
    for (int g = 0; g < 8; ++g) {
      const float kg = (float)(g + 1);
      a[g]     = (_Float16)__cosf(kg * xi);
      a[8 + g] = (_Float16)__cosf(fmaf(kg, xi, -1.57079632679489662f));
    }

    // --- 8 B fragments, each a single coalesced 32B load per lane, then WMMA.
#pragma unroll
    for (int n = 0; n < 8; ++n) {
      const size_t bidx = ((size_t)t * 512u + (size_t)(f0 + n * 16 + ln)) * 2u + (size_t)lh;
      v16h bfrag = ((const v16h*)wsB)[bidx];
      acc[n] = __builtin_amdgcn_wmma_f32_16x16x32_f16(
          false, a, false, bfrag, (short)0, acc[n], false, false);
    }

    // Prefetch next K-step's B tile (8KB) into cache: global_prefetch_b8.
    if (t + 1 < 256) {
      const _Float16* nxt = wsB + ((size_t)(t + 1) * 512u + (size_t)f0) * 32u
                                + (size_t)lane * 128u;
      __builtin_prefetch((const void*)nxt, 0, 1);
    }
  }

  // --- Store: 32-bit C/D layout: VGPR j -> (M = j + 8*lh, N = ln).
  float* yb = y + (size_t)(blockIdx.x * 128 + wave * 16 + lh * 8) * FEATURES
                + (size_t)f0 + (size_t)ln;
#pragma unroll
  for (int n = 0; n < 8; ++n) {
#pragma unroll
    for (int j = 0; j < 8; ++j) {
      yb[(size_t)j * FEATURES + (size_t)n * 16u] = acc[n][j];
    }
  }
}

// ---------------------------------------------------------------------------
extern "C" void kernel_launch(void* const* d_in, const int* in_sizes, int n_in,
                              void* d_out, int out_size, void* d_ws, size_t ws_size,
                              hipStream_t stream) {
  (void)in_sizes; (void)n_in; (void)out_size; (void)ws_size;
  const float* x  = (const float*)d_in[0];   // (4096, 512) f32
  const float* w  = (const float*)d_in[1];   // (2, 512, 512, 8) f32
  float*       y  = (float*)d_out;           // (4096, 512) f32
  _Float16*  wsB  = (_Float16*)d_ws;         // 8 MB f16 prepacked weights

  // Pass 1: 524288 chunks of 8 floats -> 2048 blocks.
  fkan_prepack_w<<<2048, 256, 0, stream>>>(w, wsB);

  // Pass 2: 128x128 output tiles -> grid (4096/128, 512/128).
  dim3 g(BATCH / 128, FEATURES / 128);
  fkan_wmma_gemm<<<g, 256, 0, stream>>>(x, wsB, y);
}
